// RandomProjectionAttention_4999341932656
// MI455X (gfx1250) — matrix operand order repaired and checked by reading, not compiled, not run on hardware
//
#include <hip/hip_runtime.h>
#include <hip/hip_bf16.h>

#define HEADS 16
#define NQ    2048
#define NKV   4096
#define HD    128
#define PD    64

typedef _Float16 half8  __attribute__((ext_vector_type(8)));
typedef _Float16 v16h   __attribute__((ext_vector_type(16)));
typedef float    v8f    __attribute__((ext_vector_type(8)));
typedef int      v8i    __attribute__((ext_vector_type(8)));

// cross the two 16-lane halves of the wave (lane i <-> lane i^16), 1 VALU op
__device__ __forceinline__ float xhalf16(float v) {
  int i = __builtin_bit_cast(int, v);
  i = __builtin_amdgcn_permlanex16(i, i, 0x76543210, 0xfedcba98, false, false);
  return __builtin_bit_cast(float, i);
}

// v_cvt_pk_rtz_f16_f32: pack two f32 into packed f16 dword (a -> low, b -> high)
__device__ __forceinline__ unsigned pk2(float a, float b) {
  auto h = __builtin_amdgcn_cvt_pkrtz(a, b);   // __fp16 ext_vector_type(2)
  return __builtin_bit_cast(unsigned, h);
}

// A-fragment loader for v_wmma_f32_16x16x32_f16.
// A k order used here (not the ISA table order): lane L holds row M=L%16; with h=L/16,
// elements 0..15 -> K = kbase + 16h + i (contiguous, matches B fragments).
__device__ __forceinline__ v16h load_afrag(const _Float16* row, int kbase, int h) {
  const _Float16* p = row + kbase + h * 16;
  half8 lo = *(const half8*)(p);
  half8 hi = *(const half8*)(p + 8);
  return __builtin_shufflevector(lo, hi, 0,1,2,3,4,5,6,7,8,9,10,11,12,13,14,15);
}

// Xp[row][p] = (sum_d X[row][d] * R[d][p]) * outScale, stored f16.
__global__ void proj_f16_kernel(const float* __restrict__ X, const float* __restrict__ R,
                                _Float16* __restrict__ Xp, int total, float outScale) {
  int id = blockIdx.x * 256 + threadIdx.x;
  if (id >= total) return;
  int row = id >> 6;          // PD == 64
  int p   = id & 63;
  const float* x = X + (size_t)row * HD;
  float s = 0.f;
#pragma unroll 8
  for (int d = 0; d < HD; ++d) s = fmaf(x[d], R[d * PD + p], s);
  Xp[id] = (_Float16)(s * outScale);
}

#define KP_STRIDE 72   // 64 + 8 halfs padding: b128 frag reads hit 16 distinct bank groups
#define VT_STRIDE 72

__global__ void __launch_bounds__(256, 1)
rpattn_kernel(const _Float16* __restrict__ Qp, const _Float16* __restrict__ Kp,
              const float* __restrict__ V, float* __restrict__ Out) {
  __shared__ __align__(16) _Float16 KpS[64 * KP_STRIDE];   // Kp chunk [kv][p]
  __shared__ __align__(16) _Float16 VtS[128 * VT_STRIDE];  // V chunk transposed [d][kv]

  const int tid  = threadIdx.x;
  const int wave = tid >> 5;
  const int lane = tid & 31;
  const int ln16 = lane & 15;
  const int hb   = lane >> 4;

  const int head  = blockIdx.x >> 4;   // 16 query blocks per head
  const int qblk  = blockIdx.x & 15;
  const int qbase = qblk * 128 + wave * 16;

  // Qp B-fragments (B layout: lane = q column ln16; K = hb*16 + i within each 32-chunk)
  const _Float16* qrow = Qp + ((size_t)(head * NQ + qbase + ln16)) * PD;
  v16h bq0 = *(const v16h*)(qrow + hb * 16);        // p = 0..31
  v16h bq1 = *(const v16h*)(qrow + 32 + hb * 16);   // p = 32..63

  // Softmax tracked in log2 domain: scores arrive pre-scaled by 0.125*log2(e)
  float mrun = -1e30f, lrun = 0.f;
  v8f oacc[8];
#pragma unroll
  for (int t = 0; t < 8; ++t) oacc[t] = (v8f){0.f,0.f,0.f,0.f,0.f,0.f,0.f,0.f};

  const _Float16* kpg = Kp + (size_t)head * NKV * PD;
  const float*    vg  = V  + (size_t)head * NKV * HD;

  for (int kvb = 0; kvb < NKV; kvb += 64) {
    // ---- stage Kp chunk (64 x 64 f16) ----
#pragma unroll
    for (int it = 0; it < 2; ++it) {
      int idx = tid + it * 256;            // 0..511 uint4s
      int row = idx >> 3, c = idx & 7;
      uint4 d = *(const uint4*)(kpg + (size_t)(kvb + row) * PD + c * 8);
      *(uint4*)(KpS + row * KP_STRIDE + c * 8) = d;
    }
    // ---- stage V chunk transposed, converted via v_cvt_pk_rtz_f16_f32 ----
#pragma unroll
    for (int it = 0; it < 4; ++it) {
      int idx = tid + it * 256;            // 0..1023
      int kv2 = idx >> 5;                  // kv pair 0..31
      int dq  = (idx & 31) << 2;           // 0..124
      const float* vp = vg + (size_t)(kvb + kv2 * 2) * HD + dq;
      float4 a = *(const float4*)(vp);
      float4 b = *(const float4*)(vp + HD);
      *(unsigned*)(VtS + (dq + 0) * VT_STRIDE + kv2 * 2) = pk2(a.x, b.x);
      *(unsigned*)(VtS + (dq + 1) * VT_STRIDE + kv2 * 2) = pk2(a.y, b.y);
      *(unsigned*)(VtS + (dq + 2) * VT_STRIDE + kv2 * 2) = pk2(a.z, b.z);
      *(unsigned*)(VtS + (dq + 3) * VT_STRIDE + kv2 * 2) = pk2(a.w, b.w);
    }
    if (kvb + 64 < NKV)
      __builtin_prefetch(vg + (size_t)(kvb + 64) * HD + lane * 128, 0, 0);
    __syncthreads();

    // ---- S^T tiles: (16 kv x 16 q) = Kp_chunk x Qp^T, 2 WMMAs (K=32+32) each ----
    float pt[4][8];
    float cmax = -1e30f;
#pragma unroll
    for (int t = 0; t < 4; ++t) {
      const _Float16* arow = KpS + (t * 16 + ln16) * KP_STRIDE;
      v16h a0 = load_afrag(arow, 0, hb);
      v16h a1 = load_afrag(arow, 32, hb);
      v8f acc = (v8f){0.f,0.f,0.f,0.f,0.f,0.f,0.f,0.f};
      acc = __builtin_amdgcn_wmma_f32_16x16x32_f16(false, a0, false, bq0, (short)0, acc, false, false);
      acc = __builtin_amdgcn_wmma_f32_16x16x32_f16(false, a1, false, bq1, (short)0, acc, false, false);
#pragma unroll
      for (int j = 0; j < 8; ++j) {
        float s = acc[j];                  // already scaled by 0.125*log2(e)
        pt[t][j] = s;
        cmax = fmaxf(cmax, s);
      }
    }
    // each lane holds 32 scores of ONE query row; other 32 are in lane^16
    cmax = fmaxf(cmax, xhalf16(cmax));
    float mnew  = fmaxf(mrun, cmax);
    float alpha = __builtin_amdgcn_exp2f(mrun - mnew);
    mrun = mnew;
    float csum = 0.f;
#pragma unroll
    for (int t = 0; t < 4; ++t)
#pragma unroll
      for (int j = 0; j < 8; ++j) {
        float e = __builtin_amdgcn_exp2f(pt[t][j] - mnew);
        pt[t][j] = e;
        csum += e;
      }
    csum += xhalf16(csum);
    lrun = lrun * alpha + csum;
#pragma unroll
    for (int t = 0; t < 8; ++t)
#pragma unroll
      for (int j = 0; j < 8; ++j) oacc[t][j] *= alpha;

    // ---- build P^T B-fragments from S^T C/D layout (packed) ----
    // B elem i of lane (ln16 + 16*hb): kv = kc*32 + hb*16 + i, tile tm = 2kc+hb.
    // My copy of tile tm holds kv rows 8*hb+j; rows 8*(1-hb)+j come from lane^16
    // (whose cross tile tc = 2kc+1-hb equals my tm).
    v16h bp[2];
#pragma unroll
    for (int kc = 0; kc < 2; ++kc) {
      const int tm = 2 * kc + hb;
      const int tc = 2 * kc + 1 - hb;
      v8i w;
#pragma unroll
      for (int j = 0; j < 8; j += 2) {
        unsigned pm = pk2(pt[tm][j], pt[tm][j + 1]);
        unsigned pc = pk2(xhalf16(pt[tc][j]), xhalf16(pt[tc][j + 1]));
        w[(j >> 1)]     = (int)((hb == 0) ? pm : pc);
        w[(j >> 1) + 4] = (int)((hb == 0) ? pc : pm);
      }
      bp[kc] = __builtin_bit_cast(v16h, w);
    }

    // ---- O^T += V^T x P^T : 8 d-tiles x (K=32+32) ----
#pragma unroll
    for (int dt = 0; dt < 8; ++dt) {
      const _Float16* arow = VtS + (dt * 16 + ln16) * VT_STRIDE;
      v16h av0 = load_afrag(arow, 0, hb);
      v16h av1 = load_afrag(arow, 32, hb);
      oacc[dt] = __builtin_amdgcn_wmma_f32_16x16x32_f16(false, av0, false, bp[0], (short)0, oacc[dt], false, false);
      oacc[dt] = __builtin_amdgcn_wmma_f32_16x16x32_f16(false, av1, false, bp[1], (short)0, oacc[dt], false, false);
    }
    __syncthreads();
  }

  // O^T C/D layout: lane column = query row (qbase+ln16); element j of d-tile dt is d = dt*16 + 8*hb + j
  float inv = 1.f / lrun;
  float* orow = Out + ((size_t)(head * NQ + qbase + ln16)) * HD;
#pragma unroll
  for (int dt = 0; dt < 8; ++dt)
#pragma unroll
    for (int j = 0; j < 8; ++j)
      orow[dt * 16 + hb * 8 + j] = oacc[dt][j] * inv;
}

extern "C" void kernel_launch(void* const* d_in, const int* in_sizes, int n_in,
                              void* d_out, int out_size, void* d_ws, size_t ws_size,
                              hipStream_t stream) {
  const float* Q = (const float*)d_in[0];
  const float* K = (const float*)d_in[1];
  const float* V = (const float*)d_in[2];
  const float* R = (const float*)d_in[3];
  float* out = (float*)d_out;

  _Float16* Qp = (_Float16*)d_ws;                       // 16*2048*64 f16 = 4 MB
  _Float16* Kp = Qp + (size_t)HEADS * NQ * PD;          // 16*4096*64 f16 = 8 MB

  const int qtot = HEADS * NQ * PD;    // 2,097,152
  const int ktot = HEADS * NKV * PD;   // 4,194,304

  // Fold softmax scale (1/sqrt(64)) AND log2(e) into Qp so the inner loop
  // needs no per-score multiply and can use raw v_exp_f32 (exp2).
  const float qScale = 0.125f * 1.4426950408889634f;
  proj_f16_kernel<<<(qtot + 255) / 256, 256, 0, stream>>>(Q, R, Qp, qtot, qScale);
  proj_f16_kernel<<<(ktot + 255) / 256, 256, 0, stream>>>(K, R, Kp, ktot, 1.0f);

  rpattn_kernel<<<HEADS * 16, 256, 0, stream>>>(Qp, Kp, V, out);
}
